// FRCRPolicy_54597624267010
// MI455X (gfx1250) — matrix operand order repaired and checked by reading, not compiled, not run on hardware
//
#include <hip/hip_runtime.h>
#include <hip/hip_bf16.h>

// ---------------------------------------------------------------------------
// CDNA5 (gfx1250) full-model forward for FRCRPolicy reference.
// Dense math: v_wmma_f32_16x16x32_f16 (wave32). Weight tiles staged to LDS
// with the Tensor Data Mover (tensor_load_to_lds + s_wait_tensorcnt),
// double-buffered so DMA overlaps WMMA.
// ---------------------------------------------------------------------------

typedef __attribute__((ext_vector_type(16))) _Float16 v16h;
typedef __attribute__((ext_vector_type(8)))  float    v8f;
typedef __attribute__((ext_vector_type(4)))  unsigned int u32x4;
typedef __attribute__((ext_vector_type(8)))  int      i32x8;
typedef __attribute__((ext_vector_type(4)))  int      i32x4;

#define BATCH 4096

static __device__ inline v8f wmma_f16(v16h a, v16h b, v8f c) {
  return __builtin_amdgcn_wmma_f32_16x16x32_f16(false, a, false, b, (short)0, c,
                                                false, false);
}

static __device__ inline float gelu_f(float x) {
  return 0.5f * x * (1.0f + erff(x * 0.70710678118654752f));
}

// ---------------------------------------------------------------------------
// TDM: 2D tile load Global f32 -> LDS (contiguous, row-major tile_y x tile_x).
// D# layout per CDNA5 ISA 8.3/8.4: group0 = {count/type/lds_addr/global_addr},
// group1 = {data_size, tensor dims, tile dims, dim0 stride}. Tile dims are set
// equal to tensor dims (caller guarantees fully in-bounds tiles).
// This toolchain exposes the 6-arg builtin:
//   (uint32x4 g0, int32x8 g1, int32x4 g2, int32x4 g3, int32x8 g4, i32 cpol)
// ---------------------------------------------------------------------------
static __device__ inline void tdm_load_2d(unsigned lds_byte_off, const float* gptr,
                                          int tile_x, int tile_y, int stride_elems) {
  unsigned long long ga = (unsigned long long)(const void*)gptr;
  u32x4 g0;
  g0[0] = 1u;                                   // count=1, user desc
  g0[1] = lds_byte_off;                         // lds_addr (bytes)
  g0[2] = (unsigned)(ga & 0xFFFFFFFFu);         // global_addr[31:0]
  g0[3] = (unsigned)((ga >> 32) & 0x1FFFFFFu)   // global_addr[56:32]
          | (2u << 30);                         // type = 2 ("image")
  i32x8 g1;
  g1[0] = 2 << 16;                              // data_size = 2 (4 bytes)
  g1[1] = (tile_x & 0xFFFF) << 16;              // tensor_dim0[15:0]
  g1[2] = ((unsigned)tile_x >> 16) | (tile_y << 16);  // dim0 hi | tensor_dim1 lo
  g1[3] = (tile_x << 16);                       // tensor_dim1 hi | tile_dim0
  g1[4] = tile_y;                               // tile_dim1 | tile_dim2(0)
  g1[5] = stride_elems;                         // tensor_dim0_stride[31:0]
  g1[6] = 0;
  g1[7] = 0;
  i32x4 z4 = {};
  i32x8 z8 = {};
  __builtin_amdgcn_tensor_load_to_lds(g0, g1, z4, z4, z8, 0);
}

template <int N>
static __device__ inline void tdm_wait() {
  __builtin_amdgcn_s_wait_tensorcnt(N);  // constant required by the builtin
}

// A fragment (16x32 f16, M x K). Lane: M = lane&15. VGPR v holds
// K = (lane/16)*16 + 2v and +1.
template <class F>
static __device__ inline v16h load_a_frag(int k0, F f) {
  int lane = threadIdx.x & 31;
  int half = lane >> 4;
  v16h a;
#pragma unroll
  for (int v = 0; v < 8; ++v) {
    int kb = k0 + half * 16 + 2 * v;
    a[2 * v]     = (_Float16)f(kb);
    a[2 * v + 1] = (_Float16)f(kb + 1);
  }
  return a;
}

// B fragment (32x16 f16, K x N). Lane: N = lane&15. VGPR v holds
// K = (lane/16)*16 + 2v and +1.
template <class F>
static __device__ inline v16h load_b_frag(int k0, F f) {
  int lane  = threadIdx.x & 31;
  int khalf = lane >> 4;
  v16h b;
#pragma unroll
  for (int v = 0; v < 8; ++v) {
    int k = k0 + khalf * 16 + 2 * v;
    b[2 * v]     = (_Float16)f(k);
    b[2 * v + 1] = (_Float16)f(k + 1);
  }
  return b;
}

// ---------------------------------------------------------------------------
// LDS-staged GEMM (requires M%64==0, N%64==0, K%32==0).
// 4 waves / block; block tile 64(M) x 64(N). Wave 0 TDM-stages the 32x64 f32
// weight tile, double-buffered (2 x 8KB LDS); all waves read B from LDS.
// C = epi(concat(A1[:, :K1], A2[:, K1:]) @ W + bias) (+ optional residual R).
// act: 0 none, 1 relu, 2 gelu.
// ---------------------------------------------------------------------------
template <typename TA>
__global__ void gemm_lds_k(const TA* __restrict__ A1, int lda1, int K1,
                           const TA* __restrict__ A2, int lda2,
                           const float* __restrict__ W, const float* __restrict__ bias,
                           const float* __restrict__ R, int ldr,
                           float* __restrict__ C, int ldc, int N, int K, int act) {
  __shared__ float wtile[2][32 * 64];
  int tid  = threadIdx.x;
  int lane = tid & 31;
  int wave = tid >> 5;
  int m0 = (blockIdx.x * 4 + wave) * 16;
  int n0 = blockIdx.y * 64;
  int arow = m0 + (lane & 15);
  int nloc = lane & 15;
  int nk = K / 32;

  auto issue = [&](int i) {
    tdm_load_2d((unsigned)((i & 1) * 32 * 64 * 4),
                W + (size_t)i * 32 * N + n0, 64, 32, N);
  };
  if (tid < 32) issue(0);

  v8f acc[4] = {};
  for (int i = 0; i < nk; ++i) {
    if (tid < 32) {
      if (i + 1 < nk) {
        issue(i + 1);
        tdm_wait<1>();  // tile i complete (TDM is in-order per wave)
      } else {
        tdm_wait<0>();
      }
    }
    __syncthreads();
    volatile const float* wb = wtile[i & 1];
    int k0 = i * 32;
    v16h a = load_a_frag(k0, [&](int kk) -> float {
      return (kk < K1) ? (float)A1[(size_t)arow * lda1 + kk]
                       : (float)A2[(size_t)arow * lda2 + (kk - K1)];
    });
#pragma unroll
    for (int t = 0; t < 4; ++t) {
      v16h b = load_b_frag(0, [&](int kk) -> float {
        return wb[kk * 64 + t * 16 + nloc];
      });
      acc[t] = wmma_f16(a, b, acc[t]);
    }
    __syncthreads();  // all waves done with buffer i before it is reissued
  }

  int mb = m0 + (lane >> 4) * 8;
#pragma unroll
  for (int t = 0; t < 4; ++t) {
    int n = n0 + t * 16 + nloc;
#pragma unroll
    for (int v = 0; v < 8; ++v) {
      int m = mb + v;
      float val = acc[t][v] + bias[n];
      if (act == 1) val = fmaxf(val, 0.0f);
      else if (act == 2) val = gelu_f(val);
      if (R) val += R[(size_t)m * ldr + n];
      C[(size_t)m * ldc + n] = val;
    }
  }
}

// ---------------------------------------------------------------------------
// Fallback generic GEMM for small/odd shapes (K<32 or N%64!=0).
// One wave per 16x16 C tile, scalar-gather fragments.
// ---------------------------------------------------------------------------
template <typename TA>
__global__ void gemm_k(const TA* __restrict__ A1, int lda1, int K1,
                       const TA* __restrict__ A2, int lda2,
                       const float* __restrict__ W, const float* __restrict__ bias,
                       const float* __restrict__ R, int ldr,
                       float* __restrict__ C, int ldc, int N, int K, int act) {
  int lane = threadIdx.x;
  int m0 = blockIdx.x * 16, n0 = blockIdx.y * 16;
  int arow = m0 + (lane & 15);
  int bcol = n0 + (lane & 15);
  v8f acc = {};
  for (int k0 = 0; k0 < K; k0 += 32) {
    v16h a = load_a_frag(k0, [&](int kk) -> float {
      if (kk >= K) return 0.0f;
      return (kk < K1) ? (float)A1[(size_t)arow * lda1 + kk]
                       : (float)A2[(size_t)arow * lda2 + (kk - K1)];
    });
    v16h b = load_b_frag(k0, [&](int kk) -> float {
      return (kk < K && bcol < N) ? W[(size_t)kk * N + bcol] : 0.0f;
    });
    acc = wmma_f16(a, b, acc);
  }
  int mb = m0 + (lane >> 4) * 8;
  int n  = n0 + (lane & 15);
#pragma unroll
  for (int v = 0; v < 8; ++v) {
    int m = mb + v;
    if (n < N) {
      float val = acc[v] + bias[n];
      if (act == 1) val = fmaxf(val, 0.0f);
      else if (act == 2) val = gelu_f(val);
      if (R) val += R[(size_t)m * ldr + n];
      C[(size_t)m * ldc + n] = val;
    }
  }
}

// ---------------------------------------------------------------------------
// Implicit-GEMM conv (3x3, stride 2, pad 1) + bias + ReLU, NCHW in/out.
// 4 waves / block (64 output pixels x 16 output channels). The 16-channel
// filter slice (16 x Cin*9 f32, <= 36KB) is TDM-staged into LDS once, then the
// whole K loop reads B fragments from LDS. Output stored f16.
// ---------------------------------------------------------------------------
template <typename TIN>
__global__ void conv_k(const TIN* __restrict__ in, const float* __restrict__ wt,
                       const float* __restrict__ bias, _Float16* __restrict__ out,
                       int Cin, int H, int Wd, int Cout, int OH, int OW, int M) {
  __shared__ float wlds[16 * 576];
  int tid  = threadIdx.x;
  int lane = tid & 31;
  int wave = tid >> 5;
  int K = Cin * 9;
  int Kpad = (K + 31) & ~31;
  int n0 = blockIdx.y * 16;

  if (tid < 32) {
    tdm_load_2d(0u, wt + (size_t)n0 * K, K, 16, K);  // 16 rows of K filters
    tdm_wait<0>();
  }
  __syncthreads();
  volatile const float* wb = wlds;

  int ohow = OH * OW;
  int m0 = (blockIdx.x * 4 + wave) * 16;
  int arow = m0 + (lane & 15);
  int bimg = arow / ohow;
  int pix  = arow - bimg * ohow;
  int oy = pix / OW, ox = pix - oy * OW;
  int iy0 = 2 * oy - 1, ix0 = 2 * ox - 1;
  int nloc = lane & 15;
  v8f acc = {};
  for (int k0 = 0; k0 < Kpad; k0 += 32) {
    v16h a = load_a_frag(k0, [&](int kk) -> float {
      if (kk >= K) return 0.0f;
      int ic = kk / 9;
      int r  = kk - ic * 9;
      int ky = r / 3, kx = r - ky * 3;
      int iy = iy0 + ky, ix = ix0 + kx;
      if ((unsigned)iy >= (unsigned)H || (unsigned)ix >= (unsigned)Wd) return 0.0f;
      return (float)in[((size_t)(bimg * Cin + ic) * H + iy) * Wd + ix];
    });
    v16h b = load_b_frag(k0, [&](int kk) -> float {
      return (kk < K) ? wb[nloc * K + kk] : 0.0f;
    });
    acc = wmma_f16(a, b, acc);
  }
  int mb = m0 + (lane >> 4) * 8;
  int n  = n0 + nloc;
#pragma unroll
  for (int v = 0; v < 8; ++v) {
    int m = mb + v;
    if (m < M) {
      float val = fmaxf(acc[v] + bias[n], 0.0f);
      int b2 = m / ohow;
      int p2 = m - b2 * ohow;
      out[(size_t)(b2 * Cout + n) * ohow + p2] = (_Float16)val;
    }
  }
}

// ---------------------------------------------------------------------------
// Similarity tile: sim[i,j] = -(max(dq,0) + 0.5 max(dp,0) + 0.5 max(dr,0)),
// dX = |x_i|^2 + |y_j|^2 - 2 x_i.y_j ; diag = -1e9. Three WMMA accumulators.
// ---------------------------------------------------------------------------
__global__ void sim_k(const float* __restrict__ q, const float* __restrict__ kk_,
                      const float* __restrict__ pe, const float* __restrict__ re,
                      const float* __restrict__ nq, const float* __restrict__ nk,
                      const float* __restrict__ nph, const float* __restrict__ nrel,
                      float* __restrict__ sim, int Bn) {
  int lane = threadIdx.x;
  int i0 = blockIdx.x * 16, j0 = blockIdx.y * 16;
  int ar = i0 + (lane & 15);
  int bc = j0 + (lane & 15);
  v8f aq = {}, ap = {}, arr = {};
  for (int k0 = 0; k0 < 256; k0 += 32) {
    v16h a = load_a_frag(k0, [&](int t) -> float { return q[(size_t)ar * 256 + t]; });
    v16h b = load_b_frag(k0, [&](int t) -> float { return kk_[(size_t)bc * 256 + t]; });
    aq = wmma_f16(a, b, aq);
  }
  for (int k0 = 0; k0 < 64; k0 += 32) {
    v16h a = load_a_frag(k0, [&](int t) -> float { return pe[(size_t)ar * 384 + t]; });
    v16h b = load_b_frag(k0, [&](int t) -> float { return pe[(size_t)bc * 384 + t]; });
    ap = wmma_f16(a, b, ap);
    a = load_a_frag(k0, [&](int t) -> float { return re[(size_t)ar * 384 + t]; });
    b = load_b_frag(k0, [&](int t) -> float { return re[(size_t)bc * 384 + t]; });
    arr = wmma_f16(a, b, arr);
  }
  int mb = i0 + (lane >> 4) * 8;
  int j  = j0 + (lane & 15);
#pragma unroll
  for (int v = 0; v < 8; ++v) {
    int i = mb + v;
    float dq = fmaxf(nq[i] + nk[j] - 2.0f * aq[v], 0.0f);
    float dp = fmaxf(nph[i] + nph[j] - 2.0f * ap[v], 0.0f);
    float dr = fmaxf(nrel[i] + nrel[j] - 2.0f * arr[v], 0.0f);
    float s = -(dq + 0.5f * dp + 0.5f * dr);
    if (i == j) s = -1.0e9f;
    sim[(size_t)i * Bn + j] = s;
  }
}

// LayerNorm, in-place, one block per row (blockDim == cols, power of two).
__global__ void ln_k(float* __restrict__ X, int ldx, int cols,
                     const float* __restrict__ g, const float* __restrict__ b) {
  __shared__ float red[256];
  int row = blockIdx.x, tid = threadIdx.x;
  float x = X[(size_t)row * ldx + tid];
  red[tid] = x;
  __syncthreads();
  for (int s = blockDim.x >> 1; s > 0; s >>= 1) {
    if (tid < s) red[tid] += red[tid + s];
    __syncthreads();
  }
  float mu = red[0] / cols;
  __syncthreads();
  float d = x - mu;
  red[tid] = d * d;
  __syncthreads();
  for (int s = blockDim.x >> 1; s > 0; s >>= 1) {
    if (tid < s) red[tid] += red[tid + s];
    __syncthreads();
  }
  float var = red[0] / cols;
  X[(size_t)row * ldx + tid] = d * rsqrtf(var + 1e-5f) * g[tid] + b[tid];
}

// Row squared-norm, one block per row (blockDim == cols).
__global__ void normsq_k(const float* __restrict__ X, int ldx,
                         float* __restrict__ out) {
  __shared__ float red[256];
  int row = blockIdx.x, tid = threadIdx.x;
  float x = X[(size_t)row * ldx + tid];
  red[tid] = x * x;
  __syncthreads();
  for (int s = blockDim.x >> 1; s > 0; s >>= 1) {
    if (tid < s) red[tid] += red[tid + s];
    __syncthreads();
  }
  if (tid == 0) out[row] = red[0];
}

// softmax(phase_logits) @ phase_emb  ->  out[row, j] (ldo stride). 64 threads.
__global__ void phase_k(const float* __restrict__ logits,
                        const float* __restrict__ emb,
                        float* __restrict__ out, int ldo) {
  int row = blockIdx.x, j = threadIdx.x;
  float l0 = logits[row * 4 + 0], l1 = logits[row * 4 + 1];
  float l2 = logits[row * 4 + 2], l3 = logits[row * 4 + 3];
  float m = fmaxf(fmaxf(l0, l1), fmaxf(l2, l3));
  float e0 = expf(l0 - m), e1 = expf(l1 - m), e2 = expf(l2 - m), e3 = expf(l3 - m);
  float Z = e0 + e1 + e2 + e3;
  float o = (e0 * emb[j] + e1 * emb[64 + j] + e2 * emb[128 + j] + e3 * emb[192 + j]) / Z;
  out[(size_t)row * ldo + j] = o;
}

// Per-row top-8: per-thread sorted insertion scan + LDS merge tree. 256 thr.
__global__ void topk_k(const float* __restrict__ sim, float* __restrict__ topv,
                       int* __restrict__ topi, int Bn) {
  __shared__ float sv[2048];
  __shared__ int   si[2048];
  int row = blockIdx.x, tid = threadIdx.x;
  float lv[8];
  int li[8];
#pragma unroll
  for (int t = 0; t < 8; ++t) { lv[t] = -3.0e38f; li[t] = 0; }
  for (int c = tid; c < Bn; c += 256) {
    float x = sim[(size_t)row * Bn + c];
    if (x > lv[7]) {
      int p = 7;
      while (p > 0 && lv[p - 1] < x) { lv[p] = lv[p - 1]; li[p] = li[p - 1]; --p; }
      lv[p] = x;
      li[p] = c;
    }
  }
#pragma unroll
  for (int t = 0; t < 8; ++t) { sv[tid * 8 + t] = lv[t]; si[tid * 8 + t] = li[t]; }
  __syncthreads();
  for (int s = 128; s > 0; s >>= 1) {
    if (tid < s) {
      for (int e = 0; e < 8; ++e) {
        float x = sv[(tid + s) * 8 + e];
        int ix = si[(tid + s) * 8 + e];
        if (x > sv[tid * 8 + 7]) {
          int p = 7;
          while (p > 0 && sv[tid * 8 + p - 1] < x) {
            sv[tid * 8 + p] = sv[tid * 8 + p - 1];
            si[tid * 8 + p] = si[tid * 8 + p - 1];
            --p;
          }
          sv[tid * 8 + p] = x;
          si[tid * 8 + p] = ix;
        }
      }
    }
    __syncthreads();
  }
  if (tid < 8) {
    topv[row * 8 + tid] = sv[tid];
    topi[row * 8 + tid] = si[tid];
  }
}

// context[row] = sum_t softmax(scores/16)_t * v[idx_t].  256 threads (j dim).
__global__ void ctx_k(const float* __restrict__ topv, const int* __restrict__ topi,
                      const float* __restrict__ v, float* __restrict__ ctx) {
  int row = blockIdx.x, j = threadIdx.x;
  float s[8];
  int id[8];
#pragma unroll
  for (int t = 0; t < 8; ++t) { s[t] = topv[row * 8 + t]; id[t] = topi[row * 8 + t]; }
  float m = s[0];  // sorted descending
  float w[8], Z = 0.0f;
#pragma unroll
  for (int t = 0; t < 8; ++t) { w[t] = expf((s[t] - m) * 0.0625f); Z += w[t]; }
  float acc = 0.0f;
#pragma unroll
  for (int t = 0; t < 8; ++t) acc += w[t] * v[(size_t)id[t] * 256 + j];
  ctx[(size_t)row * 256 + j] = acc / Z;
}

// ---------------------------------------------------------------------------
extern "C" void kernel_launch(void* const* d_in, const int* in_sizes, int n_in,
                              void* d_out, int out_size, void* d_ws, size_t ws_size,
                              hipStream_t stream) {
  (void)in_sizes; (void)n_in; (void)out_size; (void)ws_size;
  auto F = [&](int i) { return (const float*)d_in[i]; };

  const float* image = F(0);
  const float* lds_in = F(1);
  const float *c1w = F(2), *c1b = F(3), *c2w = F(4), *c2b = F(5);
  const float *c3w = F(6), *c3b = F(7), *fw = F(8), *fb = F(9);
  const float *jw1 = F(10), *jb1 = F(11), *jw2 = F(12), *jb2 = F(13);
  const float *pw1 = F(14), *pb1 = F(15), *pw2 = F(16), *pb2 = F(17);
  const float *rw1 = F(18), *rb1 = F(19), *rw2 = F(20), *rb2 = F(21);
  const float *gw1 = F(22), *gb1 = F(23), *gw2 = F(24), *gb2 = F(25);
  const float *sw1 = F(26), *sb1 = F(27), *sg1 = F(28), *sbt1 = F(29);
  const float *sw2 = F(30), *sb2 = F(31), *sg2 = F(32), *sbt2 = F(33);
  const float *phw1 = F(34), *phb1 = F(35), *phw2 = F(36), *phb2 = F(37);
  const float* phemb = F(38);
  const float *rtw = F(39), *rtb = F(40), *rtg = F(41), *rtbt = F(42);
  const float *rhw = F(43), *rhb = F(44);
  const float *rpw = F(45), *rpb = F(46), *rpg = F(47), *rpbt = F(48);
  const float *qw = F(49), *qb = F(50), *qg = F(51), *qbt = F(52);
  const float *kw = F(53), *kb_ = F(54), *kg = F(55), *kbt = F(56);
  const float *vw = F(57), *vb = F(58), *vg = F(59), *vbt = F(60);
  const float *aw1 = F(61), *ab1 = F(62), *aw2 = F(63), *ab2 = F(64);
  const float *rcw1 = F(65), *rcb1 = F(66), *rcw2 = F(67), *rcb2 = F(68);
  const float *ghw1 = F(69), *ghb1 = F(70), *ghw2 = F(71), *ghb2 = F(72);
  const float *lhw1 = F(73), *lhb1 = F(74), *lhw2 = F(75), *lhb2 = F(76);

  // ---- workspace bump allocator (~240 MB peak) ----
  size_t off = 0;
  auto alloc = [&](size_t bytes) -> char* {
    char* p = (char*)d_ws + off;
    off += (bytes + 255) & ~(size_t)255;
    return p;
  };
  const int CH = 512;  // conv batch chunk (8 passes; caps scratch)
  _Float16* c3out = (_Float16*)alloc((size_t)BATCH * 8192 * 2);        // 67 MB
  _Float16* c1c   = (_Float16*)alloc((size_t)CH * 32 * 32 * 32 * 2);   // 34 MB
  _Float16* c2c   = (_Float16*)alloc((size_t)CH * 64 * 16 * 16 * 2);   // 17 MB
  float* vis      = (float*)alloc((size_t)BATCH * 256 * 4);
  float* semb     = (float*)alloc((size_t)BATCH * 288 * 4);
  float* tmpA     = (float*)alloc((size_t)BATCH * 128 * 4);
  float* bufL1    = (float*)alloc((size_t)BATCH * 256 * 4);  // latent1 / recovered
  float* bufL2    = (float*)alloc((size_t)BATCH * 256 * 4);  // latent
  float* plog     = (float*)alloc((size_t)BATCH * 4 * 4);
  float* rel_lat  = (float*)alloc((size_t)BATCH * 256 * 4);
  float* rel_pred = (float*)alloc((size_t)BATCH * 4 * 4);
  float* retr     = (float*)alloc((size_t)BATCH * 384 * 4);  // [adapted|phase_e|rel_e]
  float* t256     = (float*)alloc((size_t)BATCH * 256 * 4);
  float* qm       = (float*)alloc((size_t)BATCH * 256 * 4);
  float* km       = (float*)alloc((size_t)BATCH * 256 * 4);
  float* vm       = (float*)alloc((size_t)BATCH * 256 * 4);
  float* nq   = (float*)alloc((size_t)BATCH * 4);
  float* nk   = (float*)alloc((size_t)BATCH * 4);
  float* nph  = (float*)alloc((size_t)BATCH * 4);
  float* nrel = (float*)alloc((size_t)BATCH * 4);
  float* sim  = (float*)alloc((size_t)BATCH * BATCH * 4);  // 67 MB
  float* topv = (float*)alloc((size_t)BATCH * 8 * 4);
  int*   topi = (int*)alloc((size_t)BATCH * 8 * 4);
  float* ctx  = (float*)alloc((size_t)BATCH * 256 * 4);
  float* out4a = (float*)alloc((size_t)BATCH * 4 * 4);

  // Small/odd shapes: scalar-gather WMMA GEMM.
  auto gemmF = [&](const float* A1, int lda1, int K1, const float* A2, int lda2,
                   const float* Wp, const float* bp, const float* Rp, int ldr,
                   float* Cp, int ldc, int M, int N, int K, int act) {
    dim3 g(M / 16, (N + 15) / 16);
    gemm_k<float><<<g, 32, 0, stream>>>(A1, lda1, K1, A2, lda2, Wp, bp, Rp, ldr,
                                        Cp, ldc, N, K, act);
  };
  // Regular shapes (M%64, N%64, K%32): TDM-staged LDS GEMM.
  auto gemmL = [&](const float* A1, int lda1, int K1, const float* A2, int lda2,
                   const float* Wp, const float* bp, const float* Rp, int ldr,
                   float* Cp, int ldc, int N, int K, int act) {
    dim3 g(BATCH / 64, N / 64);
    gemm_lds_k<float><<<g, 128, 0, stream>>>(A1, lda1, K1, A2, lda2, Wp, bp, Rp,
                                             ldr, Cp, ldc, N, K, act);
  };

  // ---- vision stack: 3 WMMA implicit-GEMM convs (f16 activations), chunked ----
  for (int c = 0; c < BATCH; c += CH) {
    conv_k<float><<<dim3(CH * 1024 / 64, 2), 128, 0, stream>>>(
        image + (size_t)c * 3 * 64 * 64, c1w, c1b, c1c, 3, 64, 64, 32, 32, 32,
        CH * 1024);
    conv_k<_Float16><<<dim3(CH * 256 / 64, 4), 128, 0, stream>>>(
        c1c, c2w, c2b, c2c, 32, 32, 32, 64, 16, 16, CH * 256);
    conv_k<_Float16><<<dim3(CH * 64 / 64, 8), 128, 0, stream>>>(
        c2c, c3w, c3b, c3out + (size_t)c * 8192, 64, 16, 16, 128, 8, 8, CH * 64);
  }
  // vis = relu(c3_flat @ fw + fb)   (K = 8192, A in f16, TDM-staged weights)
  gemm_lds_k<_Float16><<<dim3(BATCH / 64, 4), 128, 0, stream>>>(
      c3out, 8192, 8192, c3out, 8192, fw, fb, nullptr, 0, vis, 256, 256, 8192, 1);

  // ---- state encoders -> semb[4096,288] (gelu(lin(gelu(lin)))) ----
  gemmF(lds_in, 17, 7, lds_in, 17, jw1, jb1, nullptr, 0, tmpA, 128, BATCH, 128, 7, 2);
  gemmL(tmpA, 128, 128, tmpA, 128, jw2, jb2, nullptr, 0, semb + 0, 288, 128, 128, 2);
  gemmF(lds_in + 7, 17, 3, lds_in, 17, pw1, pb1, nullptr, 0, tmpA, 64, BATCH, 64, 3, 2);
  gemmL(tmpA, 64, 64, tmpA, 64, pw2, pb2, nullptr, 0, semb + 128, 288, 64, 64, 2);
  gemmF(lds_in + 10, 17, 6, lds_in, 17, rw1, rb1, nullptr, 0, tmpA, 128, BATCH, 128, 6, 2);
  gemmL(tmpA, 128, 128, tmpA, 128, rw2, rb2, nullptr, 0, semb + 192, 288, 64, 128, 2);
  gemmF(lds_in + 16, 17, 1, lds_in, 17, gw1, gb1, nullptr, 0, tmpA, 32, BATCH, 32, 1, 2);
  gemmF(tmpA, 32, 32, tmpA, 32, gw2, gb2, nullptr, 0, semb + 256, 288, BATCH, 32, 32, 2);

  // ---- shared trunk: ln(gelu(concat(vis,semb)@w1)), ln(gelu(@w2)) ----
  gemmL(vis, 256, 256, semb, 288, sw1, sb1, nullptr, 0, bufL1, 256, 256, 544, 2);
  ln_k<<<BATCH, 256, 0, stream>>>(bufL1, 256, 256, sg1, sbt1);
  gemmL(bufL1, 256, 256, bufL1, 256, sw2, sb2, nullptr, 0, bufL2, 256, 256, 256, 2);
  ln_k<<<BATCH, 256, 0, stream>>>(bufL2, 256, 256, sg2, sbt2);

  // ---- phase head -> phase_e into retr[:,256:320] ----
  gemmL(bufL2, 256, 256, bufL2, 256, phw1, phb1, nullptr, 0, tmpA, 128, 128, 256, 2);
  gemmF(tmpA, 128, 128, tmpA, 128, phw2, phb2, nullptr, 0, plog, 4, BATCH, 4, 128, 0);
  phase_k<<<BATCH, 64, 0, stream>>>(plog, phemb, retr + 256, 384);

  // ---- relation branch -> rel_e into retr[:,320:384] ----
  gemmL(bufL2, 256, 256, semb, 288, rtw, rtb, nullptr, 0, rel_lat, 256, 256, 544, 2);
  ln_k<<<BATCH, 256, 0, stream>>>(rel_lat, 256, 256, rtg, rtbt);
  gemmF(rel_lat, 256, 256, rel_lat, 256, rhw, rhb, nullptr, 0, rel_pred, 4, BATCH, 4, 256, 0);
  gemmF(rel_pred, 4, 4, rel_pred, 4, rpw, rpb, nullptr, 0, retr + 320, 384, BATCH, 64, 4, 2);
  ln_k<<<BATCH, 64, 0, stream>>>(retr + 320, 384, 64, rpg, rpbt);

  // ---- local adapter: adapted = latent + lin(gelu(lin(latent))) -> retr[:,0:256]
  gemmL(bufL2, 256, 256, bufL2, 256, aw1, ab1, nullptr, 0, t256, 256, 256, 256, 2);
  gemmL(t256, 256, 256, t256, 256, aw2, ab2, bufL2, 256, retr, 384, 256, 256, 0);

  // ---- q/k/v = proj_block(retr) ----
  gemmL(retr, 384, 384, retr, 384, qw, qb, nullptr, 0, qm, 256, 256, 384, 2);
  ln_k<<<BATCH, 256, 0, stream>>>(qm, 256, 256, qg, qbt);
  gemmL(retr, 384, 384, retr, 384, kw, kb_, nullptr, 0, km, 256, 256, 384, 2);
  ln_k<<<BATCH, 256, 0, stream>>>(km, 256, 256, kg, kbt);
  gemmL(retr, 384, 384, retr, 384, vw, vb, nullptr, 0, vm, 256, 256, 384, 2);
  ln_k<<<BATCH, 256, 0, stream>>>(vm, 256, 256, vg, vbt);

  // ---- similarity + top-8 + context ----
  normsq_k<<<BATCH, 256, 0, stream>>>(qm, 256, nq);
  normsq_k<<<BATCH, 256, 0, stream>>>(km, 256, nk);
  normsq_k<<<BATCH, 64, 0, stream>>>(retr + 256, 384, nph);
  normsq_k<<<BATCH, 64, 0, stream>>>(retr + 320, 384, nrel);
  sim_k<<<dim3(BATCH / 16, BATCH / 16), 32, 0, stream>>>(
      qm, km, retr + 256, retr + 320, nq, nk, nph, nrel, sim, BATCH);
  topk_k<<<BATCH, 256, 0, stream>>>(sim, topv, topi, BATCH);
  ctx_k<<<BATCH, 256, 0, stream>>>(topv, topi, vm, ctx);

  // ---- recovery: recovered = adapted + lin(gelu(lin(concat(adapted, ctx)))) ----
  gemmL(retr, 384, 256, ctx, 256, rcw1, rcb1, nullptr, 0, t256, 256, 256, 512, 2);
  gemmL(t256, 256, 256, t256, 256, rcw2, rcb2, retr, 384, bufL1, 256, 256, 256, 0);

  // ---- action heads: global + local MLPs, summed ----
  gemmL(bufL1, 256, 256, bufL1, 256, ghw1, ghb1, nullptr, 0, t256, 256, 256, 256, 2);
  gemmF(t256, 256, 256, t256, 256, ghw2, ghb2, nullptr, 0, out4a, 4, BATCH, 4, 256, 0);
  gemmL(bufL1, 256, 256, bufL1, 256, lhw1, lhb1, nullptr, 0, t256, 256, 256, 256, 2);
  gemmF(t256, 256, 256, t256, 256, lhw2, lhb2, out4a, 4, (float*)d_out, 4, BATCH, 4, 256, 0);
}
